// Model_39676907882844
// MI455X (gfx1250) — matrix operand order repaired and checked by reading, not compiled, not run on hardware
//
#include <hip/hip_runtime.h>

typedef __attribute__((ext_vector_type(16))) _Float16 v16h;
typedef __attribute__((ext_vector_type(8)))  _Float16 v8h;
typedef __attribute__((ext_vector_type(4)))  _Float16 v4h;
typedef __attribute__((ext_vector_type(8)))  float    v8f;
typedef __attribute__((ext_vector_type(4)))  float    v4f;
typedef unsigned int u32;
typedef __attribute__((ext_vector_type(4))) u32 u32x4;
typedef __attribute__((ext_vector_type(8))) u32 u32x8;

#define DH    64
#define BLK   64
#define NWAVE 4
#define TPB   128

// ---- Tensor Data Mover: DMA one 64x64 f32 tile (row-major, stride 64) ------
// D# per CDNA5 ISA ch.8: group0 {count, lds_addr, global_addr, type=2},
// group1 {data_size=4B, tensor 64x64, tile 64x64, dim0_stride=64}.
// 2-operand form => descriptor groups 2/3 are NULL (2-D tensor).
__device__ __forceinline__ void tdm_load_tile64x64_f32(u32 lds_off,
                                                       const float* gptr)
{
    unsigned long long ga = (unsigned long long)(size_t)gptr;
    u32x4 g0;
    u32x8 g1;
    g0[0] = 1u;                               // count=1 (valid user D#)
    g0[1] = lds_off;                          // LDS byte address
    g0[2] = (u32)ga;                          // global_addr[31:0]
    g0[3] = (u32)(ga >> 32) | (2u << 30);     // global_addr[56:32] | type=2
    g1[0] = (2u << 16);                       // wg_mask=0, data_size=2 (4B)
    g1[1] = (64u << 16);                      // tensor_dim0 = 64
    g1[2] = (64u << 16);                      // tensor_dim1 = 64
    g1[3] = (64u << 16);                      // tile_dim0   = 64
    g1[4] = 64u;                              // tile_dim1   = 64
    g1[5] = 64u;                              // tensor_dim0_stride = 64
    g1[6] = 0u;
    g1[7] = 0u;
    asm volatile("tensor_load_to_lds %0, %1" :: "s"(g0), "s"(g1) : "memory");
}

__launch_bounds__(TPB)
__global__ void fattn64_kernel(const float* __restrict__ qp,
                               const float* __restrict__ kp,
                               const float* __restrict__ vp,
                               const unsigned char* __restrict__ maskp,
                               const float* __restrict__ Wp,
                               const float* __restrict__ bp,
                               float* __restrict__ outp,
                               int Lk)
{
    __shared__ __align__(32) _Float16 Qs[64][64];         // Q' (f16)
    __shared__ __align__(32) _Float16 Ks[BLK][DH];        // K block f16, natural
    __shared__ __align__(32) _Float16 Vt[DH][BLK];        // V block f16, transposed
    __shared__ __align__(32) _Float16 Ps[NWAVE][16][BLK]; // per-wave P bounce
    __shared__ __align__(16) float    Kst[2][BLK][DH];    // TDM f32 stage (dbl buf)
    __shared__ __align__(16) float    Vst[2][BLK][DH];

    const int b    = blockIdx.x;
    const int tid  = threadIdx.x;
    const int lane = tid & 31;
    const int wave = tid >> 5;
    const int hi   = lane >> 4;
    const int ln   = lane & 15;
    const int hz   = hi * 8;

    const size_t kvbase = (size_t)b * (size_t)Lk * DH;
    const unsigned char* mrow = maskp + (size_t)b * (size_t)Lk;

    const u32 kOff[2] = { (u32)(size_t)(void*)&Kst[0][0][0],
                          (u32)(size_t)(void*)&Kst[1][0][0] };
    const u32 vOff[2] = { (u32)(size_t)(void*)&Vst[0][0][0],
                          (u32)(size_t)(void*)&Vst[1][0][0] };

    // Kick off DMA of block 0 so it overlaps the Q' GEMM prologue.
    if (wave == 0) {
        tdm_load_tile64x64_f32(kOff[0], kp + kvbase);
        tdm_load_tile64x64_f32(vOff[0], vp + kvbase);
    }

    // ---- Q' = transpose(q @ W^T + bias) / sqrt(64), f16 into LDS ----
    {
        const int d  = tid >> 1;
        const int x0 = (tid & 1) * 32;
        const float* Wd = Wp + d * DH;
        const float  bb = bp[d];
        for (int x = x0; x < x0 + 32; ++x) {
            const float* qrow = qp + ((size_t)b * DH + x) * DH;
            float acc = 0.f;
            #pragma unroll
            for (int j = 0; j < DH; ++j) acc += qrow[j] * Wd[j];
            Qs[d][x] = (_Float16)((acc + bb) * 0.125f);
        }
    }
    __syncthreads();

    // ---- per-wave A fragments of Q' (two K=32 steps) ----
    const int am = wave * 16 + ln;
    v16h aq[2];
    #pragma unroll
    for (int s = 0; s < 2; ++s) {
        v8h lo = *(const v8h*)&Qs[am][32 * s + hi * 16];
        v8h hh = *(const v8h*)&Qs[am][32 * s + hi * 16 + 8];
        aq[s] = __builtin_shufflevector(lo, hh,
                 0,1,2,3,4,5,6,7,8,9,10,11,12,13,14,15);
    }

    // ---- flash-attention running state ----
    v8f   O[4];
    float rowm[8], rowl[8];
    #pragma unroll
    for (int t = 0; t < 4; ++t)
        #pragma unroll
        for (int r = 0; r < 8; ++r) O[t][r] = 0.f;
    #pragma unroll
    for (int r = 0; r < 8; ++r) { rowm[r] = -1e30f; rowl[r] = 0.f; }

    for (int key0 = 0; key0 < Lk; key0 += BLK) {
        const int cur = (key0 >> 6) & 1;

        // ---- wave 0: DMA next block, then ensure current block landed ----
        if (wave == 0) {
            if (key0 + BLK < Lk) {
                const float* kn = kp + kvbase + (size_t)(key0 + BLK) * DH;
                const float* vn = vp + kvbase + (size_t)(key0 + BLK) * DH;
                tdm_load_tile64x64_f32(kOff[cur ^ 1], kn);
                tdm_load_tile64x64_f32(vOff[cur ^ 1], vn);
                __builtin_amdgcn_s_wait_tensorcnt(2);  // current pair done
            } else {
                __builtin_amdgcn_s_wait_tensorcnt(0);
            }
        }
        __syncthreads();   // stage[cur] visible; prev compute done (Ks/Vt free)

        // ---- convert staged f32 -> f16 compute layouts ----
        {
            const int r  = tid >> 1;
            const int f0 = (tid & 1) * 32;
            const float* ks = &Kst[cur][r][f0];
            const float* vs = &Vst[cur][r][f0];
            #pragma unroll
            for (int j = 0; j < 8; ++j) {
                v4f kf = *(const v4f*)(ks + 4 * j);
                v4h kh;
                kh[0] = (_Float16)kf[0]; kh[1] = (_Float16)kf[1];
                kh[2] = (_Float16)kf[2]; kh[3] = (_Float16)kf[3];
                *(v4h*)&Ks[r][f0 + 4 * j] = kh;
                v4f vf = *(const v4f*)(vs + 4 * j);
                Vt[f0 + 4*j + 0][r] = (_Float16)vf[0];
                Vt[f0 + 4*j + 1][r] = (_Float16)vf[1];
                Vt[f0 + 4*j + 2][r] = (_Float16)vf[2];
                Vt[f0 + 4*j + 3][r] = (_Float16)vf[3];
            }
        }
        __syncthreads();

        // ---- S = Q' @ K_blk^T : four 16x16 tiles per wave ----
        v8f S[4];
        #pragma unroll
        for (int t = 0; t < 4; ++t) {
            v16h bk0 = *(const v16h*)&Ks[t * 16 + ln][hi * 16];
            v16h bk1 = *(const v16h*)&Ks[t * 16 + ln][32 + hi * 16];
            v8f c;
            #pragma unroll
            for (int r = 0; r < 8; ++r) c[r] = 0.f;
            c = __builtin_amdgcn_wmma_f32_16x16x32_f16(
                    false, aq[0], false, bk0, (short)0, c, false, false);
            c = __builtin_amdgcn_wmma_f32_16x16x32_f16(
                    false, aq[1], false, bk1, (short)0, c, false, false);
            const float mv = (float)mrow[key0 + t * 16 + ln];
            #pragma unroll
            for (int r = 0; r < 8; ++r) S[t][r] = c[r] + mv;
        }

        // ---- online softmax (row reductions across 16-lane halves) ----
        float mx[8];
        #pragma unroll
        for (int r = 0; r < 8; ++r) {
            float v0 = fmaxf(fmaxf(S[0][r], S[1][r]), fmaxf(S[2][r], S[3][r]));
            v0 = fmaxf(v0, __shfl_xor(v0, 1));
            v0 = fmaxf(v0, __shfl_xor(v0, 2));
            v0 = fmaxf(v0, __shfl_xor(v0, 4));
            v0 = fmaxf(v0, __shfl_xor(v0, 8));
            mx[r] = v0;
        }
        float scl[8], rs[8];
        #pragma unroll
        for (int r = 0; r < 8; ++r) {
            float mn = fmaxf(rowm[r], mx[r]);
            scl[r]  = __expf(rowm[r] - mn);
            rowm[r] = mn;
            rs[r]   = 0.f;
        }
        #pragma unroll
        for (int t = 0; t < 4; ++t)
            #pragma unroll
            for (int r = 0; r < 8; ++r) {
                float p = __expf(S[t][r] - rowm[r]);
                S[t][r] = p;
                rs[r]  += p;
            }
        #pragma unroll
        for (int r = 0; r < 8; ++r) {
            rs[r] += __shfl_xor(rs[r], 1);
            rs[r] += __shfl_xor(rs[r], 2);
            rs[r] += __shfl_xor(rs[r], 4);
            rs[r] += __shfl_xor(rs[r], 8);
            rowl[r] = rowl[r] * scl[r] + rs[r];
        }
        #pragma unroll
        for (int t = 0; t < 4; ++t)
            #pragma unroll
            for (int r = 0; r < 8; ++r) O[t][r] *= scl[r];

        // ---- C-layout P -> f16 -> per-wave LDS bounce -> A-layout ----
        #pragma unroll
        for (int t = 0; t < 4; ++t)
            #pragma unroll
            for (int r = 0; r < 8; ++r)
                Ps[wave][r + 8 * hi][t * 16 + ln] = (_Float16)S[t][r];

        v16h ap[2];
        #pragma unroll
        for (int s = 0; s < 2; ++s) {
            v8h lo = *(const v8h*)&Ps[wave][ln][32 * s + hi * 16];
            v8h hh = *(const v8h*)&Ps[wave][ln][32 * s + hi * 16 + 8];
            ap[s] = __builtin_shufflevector(lo, hh,
                     0,1,2,3,4,5,6,7,8,9,10,11,12,13,14,15);
        }

        // ---- O += P @ V_blk ----
        #pragma unroll
        for (int tf = 0; tf < 4; ++tf) {
            v16h bv0 = *(const v16h*)&Vt[tf * 16 + ln][hi * 16];
            v16h bv1 = *(const v16h*)&Vt[tf * 16 + ln][32 + hi * 16];
            O[tf] = __builtin_amdgcn_wmma_f32_16x16x32_f16(
                        false, ap[0], false, bv0, (short)0, O[tf], false, false);
            O[tf] = __builtin_amdgcn_wmma_f32_16x16x32_f16(
                        false, ap[1], false, bv1, (short)0, O[tf], false, false);
        }
    }

    // ---- epilogue: divide by softmax denominator, store f32 ----
    float inv[8];
    #pragma unroll
    for (int r = 0; r < 8; ++r) inv[r] = 1.f / rowl[r];
    #pragma unroll
    for (int tf = 0; tf < 4; ++tf)
        #pragma unroll
        for (int r = 0; r < 8; ++r) {
            const int mm = wave * 16 + r + 8 * hi;
            const int ff = tf * 16 + ln;
            outp[((size_t)b * DH + mm) * DH + ff] = O[tf][r] * inv[r];
        }
}

extern "C" void kernel_launch(void* const* d_in, const int* in_sizes, int n_in,
                              void* d_out, int out_size, void* d_ws, size_t ws_size,
                              hipStream_t stream)
{
    const float*         q    = (const float*)d_in[0];
    const float*         k    = (const float*)d_in[1];
    const float*         v    = (const float*)d_in[2];
    const unsigned char* mask = (const unsigned char*)d_in[3];
    const float*         W    = (const float*)d_in[4];
    const float*         bias = (const float*)d_in[5];
    float*               out  = (float*)d_out;

    const int B  = in_sizes[0] / (DH * DH);       // 256
    const int Lk = in_sizes[1] / (B * DH);        // 4096

    fattn64_kernel<<<dim3(B), dim3(TPB), 0, stream>>>(q, k, v, mask, W, bias,
                                                      out, Lk);
}
